// CleanDITBlock_17154099380430
// MI455X (gfx1250) — matrix operand order repaired and checked by reading, not compiled, not run on hardware
//
#include <hip/hip_runtime.h>
#include <hip/hip_bf16.h>
#include <math.h>

// ---------------- problem constants (match reference) ----------------
#define BB   2
#define LL   2048
#define SS   512
#define DD   1024
#define HH   16
#define HDD  64
#define HID  4096
#define SIXD 6144
#define EPS  1e-6f

typedef __bf16 bf;
typedef __attribute__((ext_vector_type(16))) __bf16 v16bf;
typedef __attribute__((ext_vector_type(8)))  float  v8f;

union Frag {
    v16bf v;
    float4 f4[2];
    __bf16 h[16];
};

// ---- CDNA5 async-to-LDS helpers (inline asm; ASYNCcnt-tracked) ----
__device__ __forceinline__ void async_load_b128(unsigned lds_addr, const void* gaddr) {
    asm volatile("global_load_async_to_lds_b128 %0, %1, off"
                 :: "v"(lds_addr), "v"((unsigned long long)(uintptr_t)gaddr)
                 : "memory");
}
__device__ __forceinline__ void wait_async0() {
    asm volatile("s_wait_asynccnt 0x0" ::: "memory");
}
__device__ __forceinline__ void wait_ds0() {
    asm volatile("s_wait_dscnt 0x0" ::: "memory");
}

// ---------------------------------------------------------------------
// transpose + convert: dst[N,K] (bf16) = src[K,N] (f32)
__global__ void k_transpose_bf16(bf* __restrict__ dst, const float* __restrict__ src,
                                 int K, int N) {
    int tid = blockIdx.x * blockDim.x + threadIdx.x;
    if (tid >= N * K) return;
    int n = tid / K;
    int k = tid - n * K;
    dst[tid] = (bf)src[(long)k * N + n];
}

// plain convert f32 -> bf16
__global__ void k_convert_bf16(bf* __restrict__ dst, const float* __restrict__ src, int n) {
    int tid = blockIdx.x * blockDim.x + threadIdx.x;
    if (tid < n) dst[tid] = (bf)src[tid];
}

// copy f32
__global__ void k_copy_f32(float* __restrict__ dst, const float* __restrict__ src, int n) {
    int tid = blockIdx.x * blockDim.x + threadIdx.x;
    if (tid < n) dst[tid] = src[tid];
}

// mods[b, j] = timestep_emb[b,:] @ w_mod[:, j] + b_mod[j]   (tiny GEMV, fp32)
__global__ void k_mod(const float* __restrict__ ts, const float* __restrict__ wmod,
                      const float* __restrict__ bmod, float* __restrict__ mods) {
    int tid = blockIdx.x * blockDim.x + threadIdx.x;
    if (tid >= BB * SIXD) return;
    int b = tid / SIXD;
    int j = tid - b * SIXD;
    const float* t = ts + b * DD;
    float acc = bmod[j];
#pragma unroll 4
    for (int k = 0; k < DD; ++k) acc += t[k] * wmod[(long)k * SIXD + j];
    mods[tid] = acc;
}

// LayerNorm (+ optional adaLN modulate) -> bf16.  One block per row.
// scIdx < 0  => plain LN.
__global__ void k_ln_mod(const float* __restrict__ X, const float* __restrict__ mods,
                         bf* __restrict__ out, int scIdx, int shIdx) {
    __shared__ float r1[256];
    __shared__ float r2[256];
    int row = blockIdx.x;               // 0 .. B*L-1
    int b   = row / LL;
    int tid = threadIdx.x;
    const float* x = X + (long)row * DD;
    float s = 0.f, s2 = 0.f;
#pragma unroll
    for (int i = tid; i < DD; i += 256) { float v = x[i]; s += v; s2 += v * v; }
    r1[tid] = s; r2[tid] = s2;
    __syncthreads();
    for (int off = 128; off > 0; off >>= 1) {
        if (tid < off) { r1[tid] += r1[tid + off]; r2[tid] += r2[tid + off]; }
        __syncthreads();
    }
    float mu   = r1[0] * (1.0f / DD);
    float var  = r2[0] * (1.0f / DD) - mu * mu;
    float rstd = rsqrtf(var + EPS);
    bf* o = out + (long)row * DD;
#pragma unroll
    for (int i = tid; i < DD; i += 256) {
        float nh = (x[i] - mu) * rstd;
        if (scIdx >= 0) {
            float sc = mods[b * SIXD + scIdx * DD + i];
            float sh = mods[b * SIXD + shIdx * DD + i];
            nh = nh * (1.0f + sc) + sh;
        }
        o[i] = (bf)nh;
    }
}

// ---------------------------------------------------------------------
// Generic bf16 WMMA GEMM:  C[M,N] = A[M,K] @ WT[N,K]^T  (+bias) (+GELU)
// block = 256 threads (8 waves). Wave w computes rows m0+16w..+15, 4 N-tiles.
// Block tile 128(M) x 64(N). M % 128 == 0, N % 64 == 0, K % 32 == 0.
__global__ void k_gemm_bf16(const bf* __restrict__ A, const bf* __restrict__ WT,
                            const float* __restrict__ bias,
                            float* __restrict__ outF, bf* __restrict__ outB,
                            int M, int N, int K, int act) {
    int wave = threadIdx.x >> 5;
    int lane = threadIdx.x & 31;
    int m0 = blockIdx.y * 128 + wave * 16;
    int n0 = blockIdx.x * 64;

    int arow = lane & 15;
    int ak   = (lane >> 4) * 16;   // A frag: lanes 0-15 -> K 0..15; 16-31 -> K 16..31
    int bn   = lane & 15;
    int bk   = (lane >> 4) * 16;   // B frag: lanes 0-15 -> K 0..15; 16-31 -> K 16..31

    v8f acc0 = {}, acc1 = {}, acc2 = {}, acc3 = {};
    const bf* Arow = A + (long)(m0 + arow) * K;

    for (int k0 = 0; k0 < K; k0 += 32) {
        __builtin_prefetch(Arow + k0 + 64, 0, 1);
        Frag fa;
        const float4* pa = reinterpret_cast<const float4*>(Arow + k0 + ak);
        fa.f4[0] = pa[0];          // halves k0+ak .. +7
        fa.f4[1] = pa[1];          // halves k0+ak+8 .. +15
#pragma unroll
        for (int t = 0; t < 4; ++t) {
            const bf* Wr = WT + (long)(n0 + t * 16 + bn) * K + k0 + bk;
            Frag fb;
            const float4* pb = reinterpret_cast<const float4*>(Wr);
            fb.f4[0] = pb[0];
            fb.f4[1] = pb[1];
            v8f* accp = (t == 0) ? &acc0 : (t == 1) ? &acc1 : (t == 2) ? &acc2 : &acc3;
            *accp = __builtin_amdgcn_wmma_f32_16x16x32_bf16(
                false, fa.v, false, fb.v, (short)0, *accp, false, false);
        }
    }

    // epilogue: C layout -> vgpr r: lanes0-15 row r, col=lane; lanes16-31 row 8+r, col=lane-16
    int rbase = (lane < 16) ? 0 : 8;
    int col0  = n0 + (lane & 15);
#pragma unroll
    for (int t = 0; t < 4; ++t) {
        v8f a = (t == 0) ? acc0 : (t == 1) ? acc1 : (t == 2) ? acc2 : acc3;
        int col = col0 + t * 16;
        float bv = bias ? bias[col] : 0.0f;
#pragma unroll
        for (int r = 0; r < 8; ++r) {
            int row = m0 + rbase + r;
            float v = a[r] + bv;
            if (act) v = 0.5f * v * (1.0f + erff(v * 0.7071067811865476f));
            long idx = (long)row * N + col;
            if (outF) outF[idx] = v;
            else      outB[idx] = (bf)v;
        }
    }
}

// ---------------------------------------------------------------------
// In-place RoPE on Q and K (bf16, layout [B,L,H*HD]); rope_emb f32 [L,HD]
__global__ void k_rope(bf* __restrict__ Q, bf* __restrict__ Kq,
                       const float* __restrict__ rope) {
    int tid = blockIdx.x * blockDim.x + threadIdx.x;
    int total = BB * LL * HH * (HDD / 2);
    if (tid >= total) return;
    int d   = tid & 31;
    int rem = tid >> 5;
    int h   = rem % HH;  rem /= HH;
    int l   = rem % LL;
    int b   = rem / LL;
    float e1 = rope[l * HDD + d];
    float e2 = rope[l * HDD + d + 32];
    float c1 = cosf(e1), s1 = sinf(e1);
    float c2 = cosf(e2), s2 = sinf(e2);
    long base = ((long)(b * LL + l)) * DD + h * HDD;
    float q1 = (float)Q[base + d], q2 = (float)Q[base + d + 32];
    Q[base + d]      = (bf)(q1 * c1 - q2 * s1);
    Q[base + d + 32] = (bf)(q2 * c2 + q1 * s2);
    float k1 = (float)Kq[base + d], k2 = (float)Kq[base + d + 32];
    Kq[base + d]      = (bf)(k1 * c1 - k2 * s1);
    Kq[base + d + 32] = (bf)(k2 * c2 + k1 * s2);
}

// ---------------------------------------------------------------------
// Flash attention, one wave (32 threads) per (b, h, 16-query tile).
// Q/O layout [B, Lq, D] bf16 (head offset h*HD); K/V layout [B, Lk, D] bf16.
// V tiles are staged to LDS with global_load_async_to_lds_b128 (ASYNCcnt),
// overlapping the QK^T WMMAs + softmax.
__global__ void k_flash(const bf* __restrict__ Q, const bf* __restrict__ Km,
                        const bf* __restrict__ Vm, bf* __restrict__ O,
                        int Lq, int Lk, float scale) {
    __shared__ __attribute__((aligned(16))) __bf16 pls[16 * 32];   // P tile (f32->bf16)
    __shared__ __attribute__((aligned(16))) __bf16 vls[32 * 64];   // staged V tile

    int qt = blockIdx.x, h = blockIdx.y, b = blockIdx.z;
    int lane = threadIdx.x;

    const bf* qb = Q  + (long)b * Lq * DD + h * HDD;
    const bf* kb = Km + (long)b * Lk * DD + h * HDD;
    const bf* vb = Vm + (long)b * Lk * DD + h * HDD;

    int arow = lane & 15;
    int ak   = (lane >> 4) * 16;   // A-frag d offsets 0..15 / 16..31
    int bn   = lane & 15;          // B-frag N = lane&15 (both halves)
    int bk   = (lane >> 4) * 16;   // B-frag contraction half

    // Q fragments (HD=64 -> two 16x32 A frags)
    Frag qf0, qf1;
    {
        const bf* qr = qb + (long)(qt * 16 + arow) * DD;
        const float4* p0 = reinterpret_cast<const float4*>(qr + 0 + ak);
        qf0.f4[0] = p0[0]; qf0.f4[1] = p0[1];
        const float4* p1 = reinterpret_cast<const float4*>(qr + 32 + ak);
        qf1.f4[0] = p1[0]; qf1.f4[1] = p1[1];
    }

    v8f acc0 = {}, acc1 = {}, acc2 = {}, acc3 = {};
    float m_i[8], l_i[8];
#pragma unroll
    for (int r = 0; r < 8; ++r) { m_i[r] = -1e30f; l_i[r] = 0.0f; }

    for (int kt = 0; kt < Lk; kt += 32) {
        // ---- stage V tile (rows kt..kt+31, 64 cols) into LDS asynchronously.
        // 32 rows * 128B = 8 b128 instructions * 32 lanes.
        wait_ds0();   // prior iteration's LDS reads must drain before the async loads write vls
#pragma unroll
        for (int i = 0; i < 8; ++i) {
            int chunk_id = i * 32 + lane;        // 0..255
            int vrow  = chunk_id >> 3;           // 0..31
            int vchk  = chunk_id & 7;            // 16B chunk within row
            unsigned lv = (unsigned)(uintptr_t)(vls + vrow * 64 + vchk * 8);
            const void* gv = (const void*)(vb + (long)(kt + vrow) * DD + vchk * 8);
            async_load_b128(lv, gv);
        }

        // ---- scores: two 16x16 subtiles (keys kt..kt+15, kt+16..kt+31)
        v8f sa = {}, sb = {};
        {
            const bf* krow = kb + (long)(kt + bn) * DD;     // subtile A: key = kt + (lane&15)
            Frag fb0, fb1;
            const float4* q0 = reinterpret_cast<const float4*>(krow + 0  + bk);
            fb0.f4[0] = q0[0]; fb0.f4[1] = q0[1];
            const float4* q1 = reinterpret_cast<const float4*>(krow + 32 + bk);
            fb1.f4[0] = q1[0]; fb1.f4[1] = q1[1];
            sa = __builtin_amdgcn_wmma_f32_16x16x32_bf16(false, qf0.v, false, fb0.v, (short)0, sa, false, false);
            sa = __builtin_amdgcn_wmma_f32_16x16x32_bf16(false, qf1.v, false, fb1.v, (short)0, sa, false, false);
        }
        {
            const bf* krow = kb + (long)(kt + 16 + bn) * DD; // subtile B: key = kt+16+(lane&15)
            Frag fb0, fb1;
            const float4* q0 = reinterpret_cast<const float4*>(krow + 0  + bk);
            fb0.f4[0] = q0[0]; fb0.f4[1] = q0[1];
            const float4* q1 = reinterpret_cast<const float4*>(krow + 32 + bk);
            fb1.f4[0] = q1[0]; fb1.f4[1] = q1[1];
            sb = __builtin_amdgcn_wmma_f32_16x16x32_bf16(false, qf0.v, false, fb0.v, (short)0, sb, false, false);
            sb = __builtin_amdgcn_wmma_f32_16x16x32_bf16(false, qf1.v, false, fb1.v, (short)0, sb, false, false);
        }

        // ---- online softmax over the 32 keys of this step
        float corr[8];
#pragma unroll
        for (int r = 0; r < 8; ++r) {
            float va  = sa[r] * scale;
            float vb2 = sb[r] * scale;
            float t = fmaxf(va, vb2);
#pragma unroll
            for (int msk = 1; msk < 16; msk <<= 1) t = fmaxf(t, __shfl_xor(t, msk, 32));
            float mn = fmaxf(m_i[r], t);
            corr[r] = __expf(m_i[r] - mn);
            float pa = __expf(va - mn);
            float pb = __expf(vb2 - mn);
            float rs = pa + pb;
#pragma unroll
            for (int msk = 1; msk < 16; msk <<= 1) rs += __shfl_xor(rs, msk, 32);
            l_i[r] = l_i[r] * corr[r] + rs;
            m_i[r] = mn;
            int prow = ((lane < 16) ? 0 : 8) + r;
            pls[prow * 32 + bn]      = (bf)pa;
            pls[prow * 32 + 16 + bn] = (bf)pb;
        }
        __syncthreads();

        // rescale running accumulators
#pragma unroll
        for (int r = 0; r < 8; ++r) {
            acc0[r] *= corr[r]; acc1[r] *= corr[r];
            acc2[r] *= corr[r]; acc3[r] *= corr[r];
        }

        // ---- read P back in A-fragment layout (16x32 over keys)
        Frag pf;
        {
            const float4* lp = reinterpret_cast<const float4*>(pls + arow * 32 + ak);
            pf.f4[0] = lp[0];
            pf.f4[1] = lp[1];
        }

        // ---- P(16x32) @ V(32x16) per 16-wide d tile, V from staged LDS
        wait_async0();
        int vk0 = (lane >> 4) * 16;   // lanes0-15: keys kt..kt+15 ; lanes16-31: kt+16..kt+31
#pragma unroll
        for (int dt = 0; dt < 4; ++dt) {
            Frag vf;
            int vcol = dt * 16 + bn;
#pragma unroll
            for (int j = 0; j < 16; ++j)
                vf.h[j] = vls[(vk0 + j) * 64 + vcol];
            v8f* accp = (dt == 0) ? &acc0 : (dt == 1) ? &acc1 : (dt == 2) ? &acc2 : &acc3;
            *accp = __builtin_amdgcn_wmma_f32_16x16x32_bf16(
                false, pf.v, false, vf.v, (short)0, *accp, false, false);
        }
        __syncthreads();
    }

    // ---- normalize and store
    int rbase = (lane < 16) ? 0 : 8;
#pragma unroll
    for (int dt = 0; dt < 4; ++dt) {
        v8f a = (dt == 0) ? acc0 : (dt == 1) ? acc1 : (dt == 2) ? acc2 : acc3;
        int col = h * HDD + dt * 16 + bn;
#pragma unroll
        for (int r = 0; r < 8; ++r) {
            int row = qt * 16 + rbase + r;
            float o = a[r] / l_i[r];
            O[((long)b * Lq + row) * DD + col] = (bf)o;
        }
    }
}

// ---------------------------------------------------------------------
// residual adds
__global__ void k_add_gated(float* __restrict__ X, const float* __restrict__ src,
                            const float* __restrict__ mods, int gidx) {
    int i = blockIdx.x * blockDim.x + threadIdx.x;
    int n = BB * LL * DD;
    if (i >= n) return;
    int b = i / (LL * DD);
    int dcol = i % DD;
    X[i] += mods[b * SIXD + gidx * DD + dcol] * src[i];
}

__global__ void k_add(float* __restrict__ X, const float* __restrict__ src) {
    int i = blockIdx.x * blockDim.x + threadIdx.x;
    if (i < BB * LL * DD) X[i] += src[i];
}

// ---------------------------------------------------------------------
extern "C" void kernel_launch(void* const* d_in, const int* in_sizes, int n_in,
                              void* d_out, int out_size, void* d_ws, size_t ws_size,
                              hipStream_t stream) {
    const float* x     = (const float*)d_in[0];
    const float* tse   = (const float*)d_in[1];
    const float* ctx   = (const float*)d_in[2];
    const float* rope  = (const float*)d_in[3];
    const float* w_mod = (const float*)d_in[4];
    const float* b_mod = (const float*)d_in[5];
    const float* wq_s  = (const float*)d_in[6];
    const float* wk_s  = (const float*)d_in[7];
    const float* wv_s  = (const float*)d_in[8];
    const float* wo_s  = (const float*)d_in[9];
    const float* bo_s  = (const float*)d_in[10];
    const float* wq_c  = (const float*)d_in[11];
    const float* wk_c  = (const float*)d_in[12];
    const float* wv_c  = (const float*)d_in[13];
    const float* wo_c  = (const float*)d_in[14];
    const float* bo_c  = (const float*)d_in[15];
    const float* w1    = (const float*)d_in[16];
    const float* b1    = (const float*)d_in[17];
    const float* w2    = (const float*)d_in[18];
    const float* b2    = (const float*)d_in[19];
    float* xo = (float*)d_out;

    // bump allocator on workspace
    char* cur = (char*)d_ws;
    auto alloc = [&](size_t bytes) -> void* {
        void* p = (void*)cur;
        cur += (bytes + 255) & ~(size_t)255;
        return p;
    };

    bf* wqsT = (bf*)alloc((size_t)DD * DD * 2);
    bf* wksT = (bf*)alloc((size_t)DD * DD * 2);
    bf* wvsT = (bf*)alloc((size_t)DD * DD * 2);
    bf* wosT = (bf*)alloc((size_t)DD * DD * 2);
    bf* wqcT = (bf*)alloc((size_t)DD * DD * 2);
    bf* wkcT = (bf*)alloc((size_t)DD * DD * 2);
    bf* wvcT = (bf*)alloc((size_t)DD * DD * 2);
    bf* wocT = (bf*)alloc((size_t)DD * DD * 2);
    bf* w1T  = (bf*)alloc((size_t)HID * DD * 2);
    bf* w2T  = (bf*)alloc((size_t)DD * HID * 2);
    float* mods   = (float*)alloc((size_t)BB * SIXD * 4);
    bf* nx        = (bf*)alloc((size_t)BB * LL * DD * 2);
    bf* qb        = (bf*)alloc((size_t)BB * LL * DD * 2);
    bf* kbuf      = (bf*)alloc((size_t)BB * LL * DD * 2);
    bf* vbuf      = (bf*)alloc((size_t)BB * LL * DD * 2);
    bf* attn      = (bf*)alloc((size_t)BB * LL * DD * 2);
    float* tmp    = (float*)alloc((size_t)BB * LL * DD * 4);
    bf* ctxb      = (bf*)alloc((size_t)BB * SS * 1024 * 2);
    bf* kc        = (bf*)alloc((size_t)BB * SS * DD * 2);
    bf* vc        = (bf*)alloc((size_t)BB * SS * DD * 2);
    bf* hbuf      = (bf*)alloc((size_t)BB * LL * HID * 2);

    auto cdiv = [](int a, int b) { return (a + b - 1) / b; };
    auto tr = [&](bf* dst, const float* src, int K, int N) {
        int tot = K * N;
        k_transpose_bf16<<<cdiv(tot, 256), 256, 0, stream>>>(dst, src, K, N);
    };
    auto gemm = [&](const bf* A, const bf* WT, const float* bias, float* oF, bf* oB,
                    int M, int N, int K, int act) {
        dim3 g(N / 64, M / 128);
        k_gemm_bf16<<<g, 256, 0, stream>>>(A, WT, bias, oF, oB, M, N, K, act);
    };

    // weights -> bf16 [N,K]
    tr(wqsT, wq_s, DD, DD);  tr(wksT, wk_s, DD, DD);
    tr(wvsT, wv_s, DD, DD);  tr(wosT, wo_s, DD, DD);
    tr(wqcT, wq_c, DD, DD);  tr(wkcT, wk_c, 1024, DD);
    tr(wvcT, wv_c, 1024, DD); tr(wocT, wo_c, DD, DD);
    tr(w1T, w1, DD, HID);    tr(w2T, w2, HID, DD);
    k_convert_bf16<<<cdiv(BB * SS * 1024, 256), 256, 0, stream>>>(ctxb, ctx, BB * SS * 1024);

    const int NTOK = BB * LL;          // 4096 rows
    const int NCTX = BB * SS;          // 1024 rows
    const float scale = 0.125f;        // HD^-0.5

    // running residual in d_out
    k_copy_f32<<<cdiv(NTOK * DD, 256), 256, 0, stream>>>(xo, x, NTOK * DD);

    // adaLN modulation
    k_mod<<<cdiv(BB * SIXD, 256), 256, 0, stream>>>(tse, w_mod, b_mod, mods);

    // ---------------- self attention ----------------
    k_ln_mod<<<NTOK, 256, 0, stream>>>(xo, mods, nx, /*scale_msa*/1, /*shift_msa*/0);
    gemm(nx, wqsT, nullptr, nullptr, qb,   NTOK, DD, DD, 0);
    gemm(nx, wksT, nullptr, nullptr, kbuf, NTOK, DD, DD, 0);
    gemm(nx, wvsT, nullptr, nullptr, vbuf, NTOK, DD, DD, 0);
    k_rope<<<cdiv(BB * LL * HH * 32, 256), 256, 0, stream>>>(qb, kbuf, rope);
    {
        dim3 g(LL / 16, HH, BB);
        k_flash<<<g, 32, 0, stream>>>(qb, kbuf, vbuf, attn, LL, LL, scale);
    }
    gemm(attn, wosT, bo_s, tmp, nullptr, NTOK, DD, DD, 0);
    k_add_gated<<<cdiv(NTOK * DD, 256), 256, 0, stream>>>(xo, tmp, mods, /*gate_msa*/2);

    // ---------------- cross attention ----------------
    k_ln_mod<<<NTOK, 256, 0, stream>>>(xo, mods, nx, -1, -1);   // plain LN
    gemm(nx,   wqcT, nullptr, nullptr, qb, NTOK, DD, DD, 0);
    gemm(ctxb, wkcT, nullptr, nullptr, kc, NCTX, DD, 1024, 0);
    gemm(ctxb, wvcT, nullptr, nullptr, vc, NCTX, DD, 1024, 0);
    {
        dim3 g(LL / 16, HH, BB);
        k_flash<<<g, 32, 0, stream>>>(qb, kc, vc, attn, LL, SS, scale);
    }
    gemm(attn, wocT, bo_c, tmp, nullptr, NTOK, DD, DD, 0);
    k_add<<<cdiv(NTOK * DD, 256), 256, 0, stream>>>(xo, tmp);

    // ---------------- MLP ----------------
    k_ln_mod<<<NTOK, 256, 0, stream>>>(xo, mods, nx, /*scale_mlp*/4, /*shift_mlp*/3);
    gemm(nx,   w1T, b1, nullptr, hbuf, NTOK, HID, DD, /*gelu*/1);
    gemm(hbuf, w2T, b2, tmp, nullptr, NTOK, DD, HID, 0);
    k_add_gated<<<cdiv(NTOK * DD, 256), 256, 0, stream>>>(xo, tmp, mods, /*gate_mlp*/5);

    (void)in_sizes; (void)n_in; (void)out_size; (void)ws_size;
}
